// TMCAM_88407606821445
// MI455X (gfx1250) — matrix operand order repaired and checked by reading, not compiled, not run on hardware
//
#include <hip/hip_runtime.h>
#include <hip/hip_bf16.h>

typedef __attribute__((ext_vector_type(16))) _Float16 v16h;
typedef __attribute__((ext_vector_type(8)))  float    v8f;

#define NC 32      // feature channels
#define P2 9       // 3x3 patch
#define CP 288     // NC * P2

// ---------------------------------------------------------------------------
// featA = feats[:, :, center] + prev ; featB = feats[:, :, t] + prev
// feats layout: (C, T=3, h, w) -> (c*3 + t)*hw + pix
// ---------------------------------------------------------------------------
__global__ void prep_feats_kernel(const float* __restrict__ feats,
                                  const float* __restrict__ prev,
                                  float* __restrict__ featA,
                                  float* __restrict__ featB,
                                  int hw, int t) {
  int idx = blockIdx.x * blockDim.x + threadIdx.x;
  if (idx >= NC * hw) return;
  int c = idx / hw, pix = idx % hw;
  float pv = prev ? prev[idx] : 0.0f;
  featA[idx] = feats[(c * 3 + 1) * hw + pix] + pv;
  featB[idx] = feats[(c * 3 + t) * hw + pix] + pv;
}

// ---------------------------------------------------------------------------
// out[(c*9+p)*hw + pix] = zero-padded 3x3 patch gather
// ---------------------------------------------------------------------------
__global__ void unfold3_kernel(const float* __restrict__ in,
                               float* __restrict__ out, int h, int w) {
  int idx = blockIdx.x * blockDim.x + threadIdx.x;
  int hw = h * w;
  if (idx >= CP * hw) return;
  int pix = idx % hw;
  int cp = idx / hw;
  int p = cp % 9, c = cp / 9;
  int y = pix / w, x = pix % w;
  int yy = y + p / 3 - 1, xx = x + p % 3 - 1;
  float v = 0.0f;
  if (yy >= 0 && yy < h && xx >= 0 && xx < w) v = in[c * hw + yy * w + xx];
  out[cp * hw + pix] = v;
}

// per-pixel L2 norm of 288-dim column, clamped to EPS
__global__ void colnorm_kernel(const float* __restrict__ ftm,
                               float* __restrict__ nrm, int hw) {
  int pix = blockIdx.x * blockDim.x + threadIdx.x;
  if (pix >= hw) return;
  float s = 0.0f;
  for (int ch = 0; ch < CP; ++ch) {
    float v = ftm[ch * hw + pix];
    s += v * v;
  }
  nrm[pix] = fmaxf(sqrtf(s), 1e-12f);
}

// ---------------------------------------------------------------------------
// One wave32 per pixel: cosine sim over dxd displacement window of 288-dim
// descriptors, top-4 select (stable, lowest index on ties), weighted gather.
// Lane L owns channels {L, L+32, ..., L+256}. aggd stored pixel-major (hw,288).
// ---------------------------------------------------------------------------
__global__ __launch_bounds__(256) void simtopk_kernel(
    const float* __restrict__ ftU, const float* __restrict__ ftmU,
    const float* __restrict__ normTm,
    const float* __restrict__ aggW, const float* __restrict__ aggB,
    float* __restrict__ aggd, int h, int w, int d) {
  int lane = threadIdx.x & 31;
  int wid = threadIdx.x >> 5;
  int pix = blockIdx.x * 8 + wid;
  int hw = h * w;
  if (pix >= hw) return;
  int y = pix / w, x = pix % w;
  int r = d >> 1;

  float ftv[9];
  float qs = 0.0f;
#pragma unroll
  for (int j = 0; j < 9; ++j) {
    float v = ftU[(j * 32 + lane) * hw + pix];
    ftv[j] = v;
    qs += v * v;
  }
#pragma unroll
  for (int off = 16; off; off >>= 1) qs += __shfl_xor(qs, off, 32);
  float qn = fmaxf(sqrtf(qs), 1e-12f);

  float top_v[4] = {-1e30f, -1e30f, -1e30f, -1e30f};
  int top_i[4] = {0, 0, 0, 0};
  int dcount = d * d;
  for (int dd = 0; dd < dcount; ++dd) {
    int yy = y + dd / d - r, xx = x + dd % d - r;
    float sim = 0.0f;
    if (yy >= 0 && yy < h && xx >= 0 && xx < w) {
      int q = yy * w + xx;
      float dot = 0.0f;
#pragma unroll
      for (int j = 0; j < 9; ++j) dot += ftv[j] * ftmU[(j * 32 + lane) * hw + q];
#pragma unroll
      for (int off = 16; off; off >>= 1) dot += __shfl_xor(dot, off, 32);
      sim = dot / (qn * normTm[q]);
    }
    if (sim > top_v[3]) {
      int j = 3;
      while (j > 0 && sim > top_v[j - 1]) {
        top_v[j] = top_v[j - 1];
        top_i[j] = top_i[j - 1];
        --j;
      }
      top_v[j] = sim;
      top_i[j] = dd;
    }
  }

  float wk[4] = {aggW[0], aggW[1], aggW[2], aggW[3]};
  float b0 = aggB[0];
  float acc[9];
#pragma unroll
  for (int j = 0; j < 9; ++j) acc[j] = b0;
  for (int k = 0; k < 4; ++k) {
    int dd = top_i[k];
    int yy = y + dd / d - r, xx = x + dd % d - r;
    if (yy >= 0 && yy < h && xx >= 0 && xx < w) {
      int q = yy * w + xx;
#pragma unroll
      for (int j = 0; j < 9; ++j) acc[j] += wk[k] * ftmU[(j * 32 + lane) * hw + q];
    }
  }
#pragma unroll
  for (int j = 0; j < 9; ++j) aggd[(size_t)pix * CP + (j * 32 + lane)] = acc[j];
}

// out[c,pix] = sum_p aggd[pix, c*9+p] * wm[c*9+p, pix]
__global__ void combine_kernel(const float* __restrict__ aggd,
                               const float* __restrict__ wm,
                               float* __restrict__ out, int hw) {
  int idx = blockIdx.x * blockDim.x + threadIdx.x;
  if (idx >= NC * hw) return;
  int c = idx / hw, pix = idx % hw;
  float s = 0.0f;
#pragma unroll
  for (int p = 0; p < 9; ++p)
    s += aggd[(size_t)pix * CP + c * 9 + p] * wm[(size_t)(c * 9 + p) * hw + pix];
  out[idx] = s;
}

// half-pixel bilinear x2 upsample (edge clamped), (NC,h,w) -> (NC,2h,2w)
__global__ void up2_kernel(const float* __restrict__ in, float* __restrict__ out,
                           int h, int w) {
  int oh = 2 * h, ow = 2 * w;
  int idx = blockIdx.x * blockDim.x + threadIdx.x;
  if (idx >= NC * oh * ow) return;
  int c = idx / (oh * ow);
  int rem = idx % (oh * ow);
  int oy = rem / ow, ox = rem % ow;
  float iy = oy * 0.5f - 0.25f, ix = ox * 0.5f - 0.25f;
  int y0 = (int)floorf(iy), x0 = (int)floorf(ix);
  float wy = iy - (float)y0, wx = ix - (float)x0;
  int y0c = min(max(y0, 0), h - 1), y1c = min(max(y0 + 1, 0), h - 1);
  int x0c = min(max(x0, 0), w - 1), x1c = min(max(x0 + 1, 0), w - 1);
  const float* p = in + (size_t)c * h * w;
  float v00 = p[y0c * w + x0c], v01 = p[y0c * w + x1c];
  float v10 = p[y1c * w + x0c], v11 = p[y1c * w + x1c];
  out[idx] = v00 * (1.f - wy) * (1.f - wx) + v01 * (1.f - wy) * wx +
             v10 * wy * (1.f - wx) + v11 * wy * wx;
}

// pack Xt (pixel-major, K=64, f16): k<32 -> featA, else featB
__global__ void pack_x64_kernel(const float* __restrict__ featA,
                                const float* __restrict__ featB,
                                _Float16* __restrict__ Xt, int hw) {
  int idx = blockIdx.x * blockDim.x + threadIdx.x;
  if (idx >= hw * 64) return;
  int pix = idx >> 6, k = idx & 63;
  float v = (k < 32) ? featA[k * hw + pix] : featB[(k - 32) * hw + pix];
  Xt[idx] = (_Float16)v;
}

// generic K x N (f32, row-major) -> N x K (f16) transpose pack
__global__ void pack_xT_kernel(const float* __restrict__ in,
                               _Float16* __restrict__ out, int Kd, int N) {
  int idx = blockIdx.x * blockDim.x + threadIdx.x;
  if (idx >= Kd * N) return;
  int pix = idx / Kd, k = idx % Kd;
  out[idx] = (_Float16)in[(size_t)k * N + pix];
}

__global__ void f32_to_f16_kernel(const float* __restrict__ in,
                                  _Float16* __restrict__ out, int n) {
  int idx = blockIdx.x * blockDim.x + threadIdx.x;
  if (idx < n) out[idx] = (_Float16)in[idx];
}

// ---------------------------------------------------------------------------
// WMMA GEMM: out(M x N) = Wh(M x K, f16 row-major) * Xt(N x K, f16)^T + bias
// One wave32 per 16x48 output tile: the A (weight) fragment is loaded once per
// K-chunk and reused across 3 back-to-back v_wmma_f32_16x16x32_f16 issues
// (3 accumulators), tripling WMMA work per fetched operand. K in chunks of 32.
// Fragment layouts used here (A/B in contiguous k order, not the ISA table order):
//   A: lane(0-15)=row M holds K 0..15 contiguous; lanes 16-31 hold K 16..31
//   B: lane(0-15)=col N holds K 0..15 contiguous; lanes 16-31 hold K 16..31
//   C/D: VGPR v -> row v + 8*(lane>=16), col = lane&15
// ---------------------------------------------------------------------------
__global__ __launch_bounds__(32) void gemm_wmma_kernel(
    const _Float16* __restrict__ Wh, const _Float16* __restrict__ Xt,
    const float* __restrict__ bias, float* __restrict__ out,
    int M, int N, int K) {
  int mt = blockIdx.x;
  int nt0 = blockIdx.y * 3;
  int lane = threadIdx.x;
  int half = lane >> 4;
  int l15 = lane & 15;
  int row = mt * 16 + l15;
  const _Float16* wbase = Wh + (size_t)row * K + half * 16;
  const _Float16* x0 = Xt + (size_t)(nt0 * 16 + l15) * K + half * 16;
  const _Float16* x1 = x0 + (size_t)16 * K;
  const _Float16* x2 = x1 + (size_t)16 * K;
  v8f acc0 = {}, acc1 = {}, acc2 = {};
  for (int kk = 0; kk < K; kk += 32) {
    // prefetch next K-chunk of the weight row (global_prefetch_b8)
    __builtin_prefetch(wbase + kk + 32, 0, 0);
    v16h a, b0, b1, b2;
#pragma unroll
    for (int j = 0; j < 8; ++j) {
      a[j] = wbase[kk + j];
      a[j + 8] = wbase[kk + 8 + j];
    }
#pragma unroll
    for (int j = 0; j < 16; ++j) {
      b0[j] = x0[kk + j];
      b1[j] = x1[kk + j];
      b2[j] = x2[kk + j];
    }
    acc0 = __builtin_amdgcn_wmma_f32_16x16x32_f16(false, a, false, b0, (short)0,
                                                  acc0, false, false);
    acc1 = __builtin_amdgcn_wmma_f32_16x16x32_f16(false, a, false, b1, (short)0,
                                                  acc1, false, false);
    acc2 = __builtin_amdgcn_wmma_f32_16x16x32_f16(false, a, false, b2, (short)0,
                                                  acc2, false, false);
  }
#pragma unroll
  for (int v = 0; v < 8; ++v) {
    int r = mt * 16 + v + half * 8;
    float bv = bias[r];
    size_t rb = (size_t)r * N;
    out[rb + (nt0 + 0) * 16 + l15] = acc0[v] + bv;
    out[rb + (nt0 + 1) * 16 + l15] = acc1[v] + bv;
    out[rb + (nt0 + 2) * 16 + l15] = acc2[v] + bv;
  }
}

// pixel shuffle r=2: pre(128, h, w) -> out(32, 2h, 2w)
__global__ void pshuffle_kernel(const float* __restrict__ pre,
                                float* __restrict__ out, int h, int w) {
  int oh = 2 * h, ow = 2 * w;
  int idx = blockIdx.x * blockDim.x + threadIdx.x;
  if (idx >= 32 * oh * ow) return;
  int co = idx / (oh * ow);
  int rem = idx % (oh * ow);
  int oy = rem / ow, ox = rem % ow;
  int y = oy >> 1, r1 = oy & 1, x = ox >> 1, r2 = ox & 1;
  out[idx] = pre[(size_t)(co * 4 + r1 * 2 + r2) * h * w + y * w + x];
}

// ---------------------------------------------------------------------------
extern "C" void kernel_launch(void* const* d_in, const int* in_sizes, int n_in,
                              void* d_out, int out_size, void* d_ws,
                              size_t ws_size, hipStream_t stream) {
  (void)in_sizes; (void)n_in; (void)out_size; (void)ws_size;
  const float* feats_l1 = (const float*)d_in[0];
  const float* feats_l2 = (const float*)d_in[1];
  const float* feats_l3 = (const float*)d_in[2];
  const float* agg_w[3] = {(const float*)d_in[3], (const float*)d_in[7],
                           (const float*)d_in[11]};  // l3, l2, l1
  const float* agg_b[3] = {(const float*)d_in[4], (const float*)d_in[8],
                           (const float*)d_in[12]};
  const float* wp_w[3] = {(const float*)d_in[5], (const float*)d_in[9],
                          (const float*)d_in[13]};
  const float* wp_b[3] = {(const float*)d_in[6], (const float*)d_in[10],
                          (const float*)d_in[14]};
  const float* up_w = (const float*)d_in[15];
  const float* up_b = (const float*)d_in[16];

  char* base = (char*)d_ws;
  size_t off = 0;
  auto alloc = [&](size_t bytes) -> void* {
    void* p = (void*)(base + off);
    off = (off + bytes + 255) & ~(size_t)255;
    return p;
  };
  const int HWmax = 48 * 48;
  float* featA = (float*)alloc((size_t)NC * HWmax * 4);
  float* featB = (float*)alloc((size_t)NC * HWmax * 4);
  float* ftU = (float*)alloc((size_t)CP * HWmax * 4);
  float* ftmU = (float*)alloc((size_t)CP * HWmax * 4);
  float* normTm = (float*)alloc((size_t)HWmax * 4);
  float* aggd = (float*)alloc((size_t)CP * HWmax * 4);
  float* gemmO = (float*)alloc((size_t)CP * HWmax * 4);  // also final 128xHW
  _Float16* Xt = (_Float16*)alloc((size_t)HWmax * 96 * 2);
  _Float16* Wh = (_Float16*)alloc((size_t)CP * 64 * 2);  // >= 128*96
  float* rawOut = (float*)alloc((size_t)NC * HWmax * 4);
  float* prevS = (float*)alloc((size_t)3 * NC * 24 * 24 * 4);
  float* prevL = (float*)alloc((size_t)3 * NC * HWmax * 4);
  float* catB = (float*)alloc((size_t)3 * NC * HWmax * 4);

  const int Hs[3] = {12, 24, 48};
  const int Ds[3] = {3, 5, 7};
  const float* feats[3] = {feats_l3, feats_l2, feats_l1};

  for (int lvl = 0; lvl < 3; ++lvl) {
    int h = Hs[lvl], w = h, hw = h * w;
    {
      int n = CP * 64;
      f32_to_f16_kernel<<<(n + 255) / 256, 256, 0, stream>>>(wp_w[lvl], Wh, n);
    }
    for (int t = 0; t < 3; ++t) {
      const float* prev =
          (lvl == 0) ? nullptr
                     : (lvl == 1 ? prevS + (size_t)t * NC * 24 * 24
                                 : prevL + (size_t)t * NC * hw);
      prep_feats_kernel<<<(NC * hw + 255) / 256, 256, 0, stream>>>(
          feats[lvl], prev, featA, featB, hw, t);
      unfold3_kernel<<<(CP * hw + 255) / 256, 256, 0, stream>>>(featA, ftU, h, w);
      unfold3_kernel<<<(CP * hw + 255) / 256, 256, 0, stream>>>(featB, ftmU, h, w);
      colnorm_kernel<<<(hw + 255) / 256, 256, 0, stream>>>(ftmU, normTm, hw);
      simtopk_kernel<<<hw / 8, 256, 0, stream>>>(ftU, ftmU, normTm, agg_w[lvl],
                                                 agg_b[lvl], aggd, h, w, Ds[lvl]);
      pack_x64_kernel<<<(hw * 64 + 255) / 256, 256, 0, stream>>>(featA, featB,
                                                                 Xt, hw);
      dim3 g(CP / 16, hw / 48);  // 16x48 tile per wave
      gemm_wmma_kernel<<<g, 32, 0, stream>>>(Wh, Xt, wp_b[lvl], gemmO, CP, hw, 64);
      float* dst = (lvl == 2) ? (catB + (size_t)t * NC * hw) : rawOut;
      combine_kernel<<<(NC * hw + 255) / 256, 256, 0, stream>>>(aggd, gemmO, dst,
                                                                hw);
      if (lvl == 0)
        up2_kernel<<<(NC * 4 * hw + 255) / 256, 256, 0, stream>>>(
            rawOut, prevS + (size_t)t * NC * 4 * hw, h, w);
      else if (lvl == 1)
        up2_kernel<<<(NC * 4 * hw + 255) / 256, 256, 0, stream>>>(
            rawOut, prevL + (size_t)t * NC * 4 * hw, h, w);
    }
  }

  // final: concat(96) -> 128x96 GEMM -> pixel shuffle
  const int HW1 = 48 * 48;
  {
    int n = 128 * 96;
    f32_to_f16_kernel<<<(n + 255) / 256, 256, 0, stream>>>(up_w, Wh, n);
  }
  pack_xT_kernel<<<(96 * HW1 + 255) / 256, 256, 0, stream>>>(catB, Xt, 96, HW1);
  dim3 g2(128 / 16, HW1 / 48);  // 16x48 tile per wave
  gemm_wmma_kernel<<<g2, 32, 0, stream>>>(Wh, Xt, up_b, gemmO, 128, HW1, 96);
  pshuffle_kernel<<<(32 * 4 * HW1 + 255) / 256, 256, 0, stream>>>(
      gemmO, (float*)d_out, 48, 48);
}
